// ColBERT_71494025609649
// MI455X (gfx1250) — matrix operand order repaired and checked by reading, not compiled, not run on hardware
//
#include <hip/hip_runtime.h>

// ---- types ----
typedef __attribute__((ext_vector_type(16))) _Float16 v16h;
typedef __attribute__((ext_vector_type(8)))  _Float16 v8h;
typedef __attribute__((ext_vector_type(8)))  float    v8f;
typedef int v4i_g __attribute__((vector_size(16)));   // matches builtin param type

#define HDIM 768
#define DDIM 128
#define NB   64
#define LQ   32
#define LDOC 256

// ---- CDNA5 async global->LDS path (guarded; falls back to sync copies) ----
#if defined(__gfx1250__) && __has_builtin(__builtin_amdgcn_global_load_async_to_lds_b128)
#define HAVE_ASYNC_LDS 1
#else
#define HAVE_ASYNC_LDS 0
#endif

typedef __attribute__((address_space(1))) v4i_g* as1_v4i_ptr;
typedef __attribute__((address_space(3))) v4i_g* as3_v4i_ptr;

__device__ __forceinline__ void cp16_to_lds(_Float16* lds, const _Float16* g) {
#if HAVE_ASYNC_LDS
  // Flat->AS3 is low 32 bits of the flat address (ISA 10.2).
  __builtin_amdgcn_global_load_async_to_lds_b128(
      (as1_v4i_ptr)(unsigned long long)(uintptr_t)g,
      (as3_v4i_ptr)(unsigned int)(uintptr_t)lds,
      /*offset=*/0, /*cpol=*/0);
#else
  *(v8h*)lds = *(const v8h*)g;
#endif
}

#if HAVE_ASYNC_LDS
#if __has_builtin(__builtin_amdgcn_s_wait_asynccnt)
#define ASYNC_WAIT(n) __builtin_amdgcn_s_wait_asynccnt(n)
#else
#define ASYNC_WAIT(n) asm volatile("s_wait_asynccnt %0" ::"i"(n) : "memory")
#endif
#else
#define ASYNC_WAIT(n) ((void)0)
#endif

// ---------------------------------------------------------------------------
// Kernel 0: convert W (fp32 [128,768]) -> f16, and d_mask -> additive bias
// ---------------------------------------------------------------------------
__global__ __launch_bounds__(256) void prep_kernel(
    const float* __restrict__ W, const int* __restrict__ mask,
    _Float16* __restrict__ W16, float* __restrict__ maskbias) {
  int i = blockIdx.x * 256 + threadIdx.x;
  if (i < DDIM * HDIM) W16[i] = (_Float16)W[i];
  if (i < NB * LDOC)   maskbias[i] = mask[i] ? 0.0f : -1.0e30f;
}

// ---------------------------------------------------------------------------
// Kernel 1: Y[t, 0:128] = l2norm( X[t, 0:768] @ W^T ), stored f16.
// One block = 16 token rows; 8 waves x 16-col N-tiles; K-loop of 24 WMMAs.
// ---------------------------------------------------------------------------
__global__ __launch_bounds__(256) void proj_norm_kernel(
    const float* __restrict__ X, const _Float16* __restrict__ W16,
    _Float16* __restrict__ Y) {
  __shared__ _Float16 At[16 * 776];   // 16 rows x 768 (+8 pad) f16
  __shared__ float    ssq[16];

  const int t    = threadIdx.x;
  const long row0 = (long)blockIdx.x * 16;

  // --- load 16x768 fp32 tile, convert to f16 in LDS (float4 vectorized) ---
  const float4* Xv = (const float4*)(X + row0 * HDIM);
  #pragma unroll
  for (int i = 0; i < 12; ++i) {
    int f  = t + i * 256;         // float4 index within tile, 0..3071
    int r  = f / 192;             // 192 float4 per row
    int c4 = f % 192;
    float4 v = Xv[f];
    int o = r * 776 + c4 * 4;
    At[o + 0] = (_Float16)v.x;
    At[o + 1] = (_Float16)v.y;
    At[o + 2] = (_Float16)v.z;
    At[o + 3] = (_Float16)v.w;
  }
  if (t < 16) ssq[t] = 0.0f;

  const int w  = t >> 5;          // wave id 0..7 -> N-tile
  const int l  = t & 31;          // lane
  const int nl = l & 15;          // A-row (M) and B-col (N) within tile
  const int hh = l >> 4;          // half-wave selector

  const _Float16* Wrow = W16 + (size_t)(w * 16 + nl) * HDIM;
  __builtin_prefetch(Wrow, 0, 1);   // global_prefetch_b8 on this lane's W row
  __syncthreads();

  v8f acc = {};
  for (int kk = 0; kk < 24; ++kk) {
    // A fragment (16x32 f16): two contiguous 8-half chunks per lane
    int ka = kk * 32 + hh * 8;
    v8h a0 = *(const v8h*)&At[nl * 776 + ka];
    v8h a1 = *(const v8h*)&At[nl * 776 + ka + 16];
    v16h a;
    #pragma unroll
    for (int j = 0; j < 8; ++j) { a[j] = a0[j]; a[j + 8] = a1[j]; }
    // B fragment (32x16 f16): two contiguous 8-half chunks per lane
    int kb = kk * 32 + hh * 8;
    v8h b0 = *(const v8h*)&Wrow[kb];
    v8h b1 = *(const v8h*)&Wrow[kb + 16];
    v16h bb;
    #pragma unroll
    for (int j = 0; j < 8; ++j) { bb[j] = b0[j]; bb[j + 8] = b1[j]; }
    acc = __builtin_amdgcn_wmma_f32_16x16x32_f16(
        false, a, false, bb, (short)0, acc, false, false);
  }

  // --- row sum-of-squares: reduce 16 lanes (same M), accumulate across waves
  #pragma unroll
  for (int r = 0; r < 8; ++r) {
    float sv = acc[r] * acc[r];
    sv += __shfl_xor(sv, 1, 32);
    sv += __shfl_xor(sv, 2, 32);
    sv += __shfl_xor(sv, 4, 32);
    sv += __shfl_xor(sv, 8, 32);
    if (nl == 0) atomicAdd(&ssq[r + hh * 8], sv);
  }
  __syncthreads();

  // --- scale by 1/max(||x||, eps), store normalized f16 ---
  #pragma unroll
  for (int r = 0; r < 8; ++r) {
    int m = r + hh * 8;
    float inv = 1.0f / fmaxf(sqrtf(ssq[m]), 1e-12f);
    Y[(row0 + m) * DDIM + w * 16 + nl] = (_Float16)(acc[r] * inv);
  }
}

// ---------------------------------------------------------------------------
// Kernel 2: out[b,c] = sum_q max_n ( Qn[b,q,:] . Dn[c,n,:] + bias[c,n] )
// One block per (b,c). Double-buffered 128-token D halves staged in LDS via
// GLOBAL_LOAD_ASYNC_TO_LDS_B128 (ASYNCcnt-tracked), consumed by WMMA.
// Dynamic LDS: Qt 8704B | Dt0 34816B | Dt1 34816B | pmax 1024B = 79360B.
// ---------------------------------------------------------------------------
__global__ __launch_bounds__(256) void maxsim_kernel(
    const _Float16* __restrict__ Qn, const _Float16* __restrict__ Dn,
    const float* __restrict__ maskbias, float* __restrict__ out) {
  extern __shared__ char smem[];
  _Float16* Qt   = (_Float16*)(smem);                     // 32  x 136
  _Float16* Dt0  = (_Float16*)(smem + 8704);              // 128 x 136
  _Float16* Dt1  = (_Float16*)(smem + 8704 + 34816);      // 128 x 136
  float*    pmax = (float*)(smem + 8704 + 2 * 34816);     // [32][8]

  const int t = threadIdx.x;
  const int c = blockIdx.x;
  const int b = blockIdx.y;
  const _Float16* Qg = Qn + (size_t)b * LQ * DDIM;
  const _Float16* Dg = Dn + (size_t)c * LDOC * DDIM;

  // --- issue Q tile (2 chunks/thread), then both D halves (8 each) ---
  #pragma unroll
  for (int i = 0; i < 2; ++i) {
    int idx = t + i * 256;            // 0..511
    int r = idx >> 4, c8 = idx & 15;
    cp16_to_lds(&Qt[r * 136 + c8 * 8], Qg + r * DDIM + c8 * 8);
  }
  #pragma unroll
  for (int i = 0; i < 8; ++i) {
    int idx = t + i * 256;            // 0..2047
    int r = idx >> 4, c8 = idx & 15;
    cp16_to_lds(&Dt0[r * 136 + c8 * 8], Dg + r * DDIM + c8 * 8);
  }
  #pragma unroll
  for (int i = 0; i < 8; ++i) {
    int idx = t + i * 256;
    int r = idx >> 4, c8 = idx & 15;
    cp16_to_lds(&Dt1[r * 136 + c8 * 8], Dg + (128 + r) * DDIM + c8 * 8);
  }

  const int w  = t >> 5;
  const int l  = t & 31;
  const int nl = l & 15;
  const int hh = l >> 4;

  float vmax[2][8];
  #pragma unroll
  for (int qt = 0; qt < 2; ++qt)
    #pragma unroll
    for (int r = 0; r < 8; ++r) vmax[qt][r] = -3.0e38f;

  for (int h = 0; h < 2; ++h) {
    // Async loads complete in issue order: waiting to <=8 outstanding
    // guarantees Q(2)+D0(8) landed while D1(8) may still be in flight.
    if (h == 0) { ASYNC_WAIT(8); } else { ASYNC_WAIT(0); }
    __syncthreads();

    const _Float16* Dt = h ? Dt1 : Dt0;
    float bias = maskbias[c * LDOC + h * 128 + w * 16 + nl];
    const _Float16* drow = &Dt[(w * 16 + nl) * 136];

    #pragma unroll
    for (int qt = 0; qt < 2; ++qt) {
      const _Float16* qrow = &Qt[(qt * 16 + nl) * 136];
      v8f acc = {};
      #pragma unroll
      for (int kk = 0; kk < 4; ++kk) {
        int ka = kk * 32 + hh * 8;
        v8h a0 = *(const v8h*)&qrow[ka];
        v8h a1 = *(const v8h*)&qrow[ka + 16];
        v16h a;
        #pragma unroll
        for (int j = 0; j < 8; ++j) { a[j] = a0[j]; a[j + 8] = a1[j]; }
        int kb = kk * 32 + hh * 8;
        v8h b0 = *(const v8h*)&drow[kb];
        v8h b1 = *(const v8h*)&drow[kb + 16];
        v16h bb;
        #pragma unroll
        for (int j = 0; j < 8; ++j) { bb[j] = b0[j]; bb[j + 8] = b1[j]; }
        acc = __builtin_amdgcn_wmma_f32_16x16x32_f16(
            false, a, false, bb, (short)0, acc, false, false);
      }
      #pragma unroll
      for (int r = 0; r < 8; ++r)
        vmax[qt][r] = fmaxf(vmax[qt][r], acc[r] + bias);
    }
  }

  // --- reduce max over the 16 lanes sharing each q row; write partials ---
  #pragma unroll
  for (int qt = 0; qt < 2; ++qt)
    #pragma unroll
    for (int r = 0; r < 8; ++r) {
      float m = vmax[qt][r];
      m = fmaxf(m, __shfl_xor(m, 1, 32));
      m = fmaxf(m, __shfl_xor(m, 2, 32));
      m = fmaxf(m, __shfl_xor(m, 4, 32));
      m = fmaxf(m, __shfl_xor(m, 8, 32));
      if (nl == 0) pmax[(qt * 16 + r + hh * 8) * 8 + w] = m;
    }
  __syncthreads();

  // --- final: max over 8 waves per q, sum over 32 q, store out[b,c] ---
  if (t < 32) {
    float m = pmax[t * 8];
    #pragma unroll
    for (int w2 = 1; w2 < 8; ++w2) m = fmaxf(m, pmax[t * 8 + w2]);
    m += __shfl_xor(m, 1, 32);
    m += __shfl_xor(m, 2, 32);
    m += __shfl_xor(m, 4, 32);
    m += __shfl_xor(m, 8, 32);
    m += __shfl_xor(m, 16, 32);
    if (t == 0) out[b * NB + c] = m;
  }
}

// ---------------------------------------------------------------------------
// Workspace layout (bytes):
//   [0, 196608)          W16      f16 [128][768]
//   [196608, 262144)     maskbias f32 [64*256]
//   [262144, 786432)     Qn       f16 [2048][128]
//   [786432, 4980736)    Dn       f16 [16384][128]
// ---------------------------------------------------------------------------
extern "C" void kernel_launch(void* const* d_in, const int* in_sizes, int n_in,
                              void* d_out, int out_size, void* d_ws, size_t ws_size,
                              hipStream_t stream) {
  const float* Qh   = (const float*)d_in[0];
  const float* Dh   = (const float*)d_in[1];
  const float* W    = (const float*)d_in[2];
  const int*   mask = (const int*)d_in[3];
  float* out = (float*)d_out;

  char* ws = (char*)d_ws;
  _Float16* W16      = (_Float16*)(ws);
  float*    maskbias = (float*)(ws + 196608);
  _Float16* Qn       = (_Float16*)(ws + 262144);
  _Float16* Dn       = (_Float16*)(ws + 786432);

  prep_kernel<<<384, 256, 0, stream>>>(W, mask, W16, maskbias);
  proj_norm_kernel<<<(NB * LQ) / 16, 256, 0, stream>>>(Qh, W16, Qn);     // 128 blocks
  proj_norm_kernel<<<(NB * LDOC) / 16, 256, 0, stream>>>(Dh, W16, Dn);   // 1024 blocks
  maxsim_kernel<<<dim3(NB, NB), 256, 79360, stream>>>(Qn, Dn, maskbias, out);
}
